// _EGNNLayer_14551349199304
// MI455X (gfx1250) — matrix operand order repaired and checked by reading, not compiled, not run on hardware
//
#include <hip/hip_runtime.h>

// ---------------------------------------------------------------------------
// EGNN layer for MI455X (gfx1250): fused edge pipeline with bf16 WMMA.
// B=4, N=512, H=EH=128.
// ---------------------------------------------------------------------------

typedef __attribute__((ext_vector_type(8)))  __bf16 v8bf;
typedef __attribute__((ext_vector_type(16))) __bf16 v16bf;
typedef __attribute__((ext_vector_type(8)))  float  v8f;

#define BB 4
#define NN 512
#define HH 128
#define EHH 128
#define PITCH 136      // 128 bf16 + 8 pad -> row stride 68 dwords -> 4-bank shift
#define HCPITCH 264    // 256 bf16 + 8 pad

static __device__ __forceinline__ __bf16 f2bf(float f) {
  unsigned u = __builtin_bit_cast(unsigned, f);
  unsigned r = (u + 0x7FFFu + ((u >> 16) & 1u)) >> 16;
  unsigned short s = (unsigned short)r;
  return __builtin_bit_cast(__bf16, s);
}

// Fast silu: x * v_rcp_f32(1 + exp(-x)).  Avoids the IEEE-correct divide
// (v_div_scale/v_div_fmas/v_div_fixup chains) the / operator generates.
static __device__ __forceinline__ float silu(float v) {
  return v * __builtin_amdgcn_rcpf(1.0f + __expf(-v));
}

static __device__ __forceinline__ v16bf make16(v8bf lo, v8bf hi) {
  v16bf a;
#pragma unroll
  for (int i = 0; i < 8; ++i) { a[i] = lo[i]; a[8 + i] = hi[i]; }
  return a;
}

// A fragment (16x32 bf16): lane L -> row M = L%16,
// K chunk0 = [(L<16?0:8)+32*ks, +8), chunk1 = chunk0 + 16.
static __device__ __forceinline__ v16bf load_afrag(const __bf16* p0, int pitch,
                                                   int lane, int ks) {
  int m  = lane & 15;
  int kb = ((lane >> 4) << 3) + (ks << 5);
  const __bf16* p = p0 + m * pitch + kb;
  v8bf lo = *reinterpret_cast<const v8bf*>(p);
  v8bf hi = *reinterpret_cast<const v8bf*>(p + 16);
  return make16(lo, hi);
}

// B fragment (32x16 bf16), weights stored transposed [n][k] row-major:
// lane L -> column N = n0 + L%16, K chunk0 = [(L<16?0:8)+32*ks, +8), chunk1 = chunk0 + 16.
static __device__ __forceinline__ v16bf load_bfrag(const __bf16* w, int kpitch,
                                                   int lane, int n0, int ks) {
  int n  = n0 + (lane & 15);
  int kb = ((lane >> 4) << 3) + (ks << 5);
  const __bf16* p = w + n * kpitch + kb;
  v8bf lo = *reinterpret_cast<const v8bf*>(p);
  v8bf hi = *reinterpret_cast<const v8bf*>(p + 16);
  return make16(lo, hi);
}

static __device__ __forceinline__ v8f wmma_bf16(v16bf a, v16bf b, v8f c) {
  return __builtin_amdgcn_wmma_f32_16x16x32_bf16(false, a, false, b,
                                                 (short)0, c, false, false);
}

// ---------------------------------------------------------------------------
// Precompute: Ai = h @ We1[:H] + be1 ; Bj = h @ We1[H:2H]     (f32, tiny GEMM)
// ---------------------------------------------------------------------------
__global__ __launch_bounds__(128)
void egnn_pre_embed(const float* __restrict__ h, const float* __restrict__ We1,
                    const float* __restrict__ be1,
                    float* __restrict__ Ai, float* __restrict__ Bj) {
  int row = blockIdx.x;          // b*N + n : 0..2047
  int e   = threadIdx.x;         // 0..127
  float accA = be1[e];
  float accB = 0.0f;
  const float* hr = h + row * HH;
  for (int k = 0; k < HH; ++k) {
    float hv = hr[k];
    accA = fmaf(hv, We1[k * EHH + e], accA);
    accB = fmaf(hv, We1[(HH + k) * EHH + e], accB);
  }
  Ai[row * EHH + e] = accA;
  Bj[row * EHH + e] = accB;
}

// ---------------------------------------------------------------------------
// Pack weights to bf16, transposed [n][k] for WMMA B fragments.
// ---------------------------------------------------------------------------
__global__ __launch_bounds__(256)
void egnn_pack_weights(const float* __restrict__ We2, const float* __restrict__ Wc1,
                       const float* __restrict__ Wn1, const float* __restrict__ Wn2,
                       __bf16* __restrict__ We2T, __bf16* __restrict__ Wc1T,
                       __bf16* __restrict__ Wn1T, __bf16* __restrict__ Wn2T) {
  int idx = blockIdx.x * blockDim.x + threadIdx.x;
  if (idx < 16384) {
    int n = idx >> 7, k = idx & 127;
    We2T[n * 128 + k] = f2bf(We2[k * 128 + n]);
  } else if (idx < 32768) {
    int t = idx - 16384; int n = t >> 7, k = t & 127;
    Wc1T[n * 128 + k] = f2bf(Wc1[k * 128 + n]);
  } else if (idx < 65536) {
    int t = idx - 32768; int n = t >> 8, k = t & 255;
    Wn1T[n * 256 + k] = f2bf(Wn1[k * 128 + n]);
  } else if (idx < 81920) {
    int t = idx - 65536; int n = t >> 7, k = t & 127;
    Wn2T[n * 128 + k] = f2bf(Wn2[k * 128 + n]);
  }
}

// ---------------------------------------------------------------------------
// Main fused kernel: one workgroup per (batch b, 16-node i-tile).
// Loops over 8-node j-tiles: 128 edges (M=128) per step, all in LDS.
// ---------------------------------------------------------------------------
__global__ __launch_bounds__(256)
void egnn_edge(const float* __restrict__ h, const float* __restrict__ x,
               const float* __restrict__ We1, const float* __restrict__ be2,
               const float* __restrict__ bc1, const float* __restrict__ Wc2,
               const float* __restrict__ bc2, const float* __restrict__ bn1,
               const float* __restrict__ bn2,
               const float* __restrict__ Ai, const float* __restrict__ Bj,
               const __bf16* __restrict__ We2T, const __bf16* __restrict__ Wc1T,
               const __bf16* __restrict__ Wn1T, const __bf16* __restrict__ Wn2T,
               const int* __restrict__ radius,
               float* __restrict__ outH, float* __restrict__ outX) {
  __shared__ __bf16 eBuf[128 * PITCH];     // silu(pre), A-matrix of GEMM1
  __shared__ __bf16 mBuf[128 * PITCH];     // masked m_ij (bf16), A of GEMM2
  __shared__ __bf16 hcatBuf[16 * HCPITCH]; // [h, agg] for node MLP
  __shared__ __bf16 t1Buf[16 * PITCH];     // silu(hcat@Wn1+bn1)
  __shared__ float  AiT[16 * EHH];
  __shared__ float  BjT[8 * EHH];
  __shared__ float  aggS[16 * HH];
  __shared__ float  dxBuf[128 * 3];
  __shared__ float  d2Buf[128];
  __shared__ float  maskBuf[128];
  __shared__ float  gateBuf[128];
  __shared__ float  coordAcc[16 * 3];
  __shared__ float  neighS[16];
  __shared__ float  w2hs[EHH];             // We1 row 2H (d^2 weight)
  __shared__ float  wc2s[EHH];

  const int tid  = threadIdx.x;
  const int lane = tid & 31;
  const int wave = tid >> 5;               // 0..7
  const int n0   = wave << 4;              // N-slice of EH/H
  const int b    = blockIdx.x >> 5;        // 0..3
  const int i0   = (blockIdx.x & 31) << 4; // 0..496
  const float rad = (float)(*radius);
  const float r2  = rad * rad;

  // ---- setup ----
  for (int t = tid; t < 16 * EHH / 4; t += 256) {   // float4 loads
    reinterpret_cast<float4*>(AiT)[t] =
        reinterpret_cast<const float4*>(Ai + ((b * NN) + i0) * EHH)[t];
  }
  for (int t = tid; t < 16 * HH; t += 256) aggS[t] = 0.0f;
  for (int t = tid; t < EHH; t += 256) {
    w2hs[t] = We1[2 * HH * EHH + t];
    wc2s[t] = Wc2[t];
  }
  if (tid < 48) coordAcc[tid] = 0.0f;
  if (tid < 16) neighS[tid] = 0.0f;

  // Preload B fragments (weights) for both edge GEMMs into VGPRs; they are
  // reused for all 64 j-steps.
  v16bf bw1[4], bw2[4];
#pragma unroll
  for (int ks = 0; ks < 4; ++ks) {
    bw1[ks] = load_bfrag(We2T, 128, lane, n0, ks);
    bw2[ks] = load_bfrag(Wc1T, 128, lane, n0, ks);
  }
  const int   nloc = n0 + (lane & 15);
  const float be2v = be2[nloc];
  const float bc1v = bc1[nloc];
  const float bc2v = bc2[0];
  const int   mh   = (lane >> 4) << 3;     // 0 or 8: D-fragment row half
  __syncthreads();

  // ---- j loop: 8 neighbor nodes per step -> 16x8 = 128 edges (M=128) ----
  for (int j0 = 0; j0 < NN; j0 += 8) {
    // geometry + mask + gate init
    if (tid < 128) {
      int m = tid, il = m >> 3, jl = m & 7;
      int gi = i0 + il, gj = j0 + jl;
      float dx0 = x[(b * NN + gi) * 3 + 0] - x[(b * NN + gj) * 3 + 0];
      float dx1 = x[(b * NN + gi) * 3 + 1] - x[(b * NN + gj) * 3 + 1];
      float dx2 = x[(b * NN + gi) * 3 + 2] - x[(b * NN + gj) * 3 + 2];
      float d2 = dx0 * dx0 + dx1 * dx1 + dx2 * dx2;
      dxBuf[m * 3 + 0] = dx0;
      dxBuf[m * 3 + 1] = dx1;
      dxBuf[m * 3 + 2] = dx2;
      d2Buf[m]   = d2;
      maskBuf[m] = ((d2 < r2) && (gi != gj)) ? 1.0f : 0.0f;
      gateBuf[m] = bc2v;
    }
    for (int t = tid; t < 8 * EHH / 4; t += 256) {   // float4 loads
      reinterpret_cast<float4*>(BjT)[t] =
          reinterpret_cast<const float4*>(Bj + ((b * NN) + j0) * EHH)[t];
    }
    __syncthreads();

    // eBuf = silu(Ai + Bj + d2 * We1[2H]); 8 bf16 packed per ds_store_b128
    {
      int m = tid >> 1, kh = tid & 1;
      int il = m >> 3, jl = m & 7;
      float d2v = d2Buf[m];
      const float4* ai4 = reinterpret_cast<const float4*>(&AiT[il * EHH]);
      const float4* bj4 = reinterpret_cast<const float4*>(&BjT[jl * EHH]);
      const float4* w4  = reinterpret_cast<const float4*>(w2hs);
      __bf16* er = &eBuf[m * PITCH];
#pragma unroll
      for (int kb = kh * 16; kb < kh * 16 + 16; kb += 2) {  // kb: float4 index
        float4 a0 = ai4[kb],     a1 = ai4[kb + 1];
        float4 b0 = bj4[kb],     b1 = bj4[kb + 1];
        float4 w0 = w4[kb],      w1 = w4[kb + 1];
        v8bf pk;
        pk[0] = f2bf(silu(a0.x + b0.x + d2v * w0.x));
        pk[1] = f2bf(silu(a0.y + b0.y + d2v * w0.y));
        pk[2] = f2bf(silu(a0.z + b0.z + d2v * w0.z));
        pk[3] = f2bf(silu(a0.w + b0.w + d2v * w0.w));
        pk[4] = f2bf(silu(a1.x + b1.x + d2v * w1.x));
        pk[5] = f2bf(silu(a1.y + b1.y + d2v * w1.y));
        pk[6] = f2bf(silu(a1.z + b1.z + d2v * w1.z));
        pk[7] = f2bf(silu(a1.w + b1.w + d2v * w1.w));
        *reinterpret_cast<v8bf*>(&er[kb * 4]) = pk;
      }
    }
    __syncthreads();

    // GEMM1: m = silu(e @ We2 + be2) * mask ; accumulate agg
    for (int mt = 0; mt < 8; ++mt) {
      v16bf a[4];
#pragma unroll
      for (int ks = 0; ks < 4; ++ks)
        a[ks] = load_afrag(&eBuf[(mt * 16) * PITCH], PITCH, lane, ks);
      v8f c;
#pragma unroll
      for (int r = 0; r < 8; ++r) c[r] = be2v;
#pragma unroll
      for (int ks = 0; ks < 4; ++ks) c = wmma_bf16(a[ks], bw1[ks], c);
      float aggp = 0.0f;
#pragma unroll
      for (int r = 0; r < 8; ++r) {
        int m = mt * 16 + mh + r;       // rows of this half map to one i each
        float v = silu(c[r]) * maskBuf[m];
        mBuf[m * PITCH + nloc] = f2bf(v);
        aggp += v;
      }
      atomicAdd(&aggS[(2 * mt + (lane >> 4)) * HH + nloc], aggp);
    }
    __syncthreads();

    // GEMM2: gate += silu(m @ Wc1 + bc1) . Wc2
    for (int mt = 0; mt < 8; ++mt) {
      v16bf a[4];
#pragma unroll
      for (int ks = 0; ks < 4; ++ks)
        a[ks] = load_afrag(&mBuf[(mt * 16) * PITCH], PITCH, lane, ks);
      v8f c;
#pragma unroll
      for (int r = 0; r < 8; ++r) c[r] = bc1v;
#pragma unroll
      for (int ks = 0; ks < 4; ++ks) c = wmma_bf16(a[ks], bw2[ks], c);
      float wn = wc2s[nloc];
#pragma unroll
      for (int r = 0; r < 8; ++r) {
        int m = mt * 16 + mh + r;
        atomicAdd(&gateBuf[m], silu(c[r]) * wn);
      }
    }
    __syncthreads();

    // coord delta / neighbor count
    if (tid < 128) {
      int m = tid, il = m >> 3;
      float g = gateBuf[m] * maskBuf[m];
      atomicAdd(&coordAcc[il * 3 + 0], dxBuf[m * 3 + 0] * g);
      atomicAdd(&coordAcc[il * 3 + 1], dxBuf[m * 3 + 1] * g);
      atomicAdd(&coordAcc[il * 3 + 2], dxBuf[m * 3 + 2] * g);
      atomicAdd(&neighS[il], maskBuf[m]);
    }
    __syncthreads();
  }

  // ---- node MLP: h_new = h + silu([h,agg]@Wn1+bn1)@Wn2 + bn2 ----
  for (int t = tid; t < 16 * 256; t += 256) {
    int row = t >> 8, k = t & 255;
    float v = (k < HH) ? h[((b * NN) + i0 + row) * HH + k]
                       : aggS[row * HH + (k - HH)];
    hcatBuf[row * HCPITCH + k] = f2bf(v);
  }
  __syncthreads();
  {
    v8f c;
    float bv = bn1[nloc];
#pragma unroll
    for (int r = 0; r < 8; ++r) c[r] = bv;
#pragma unroll
    for (int ks = 0; ks < 8; ++ks) {          // K = 256
      v16bf a  = load_afrag(hcatBuf, HCPITCH, lane, ks);
      v16bf bw = load_bfrag(Wn1T, 256, lane, n0, ks);
      c = wmma_bf16(a, bw, c);
    }
#pragma unroll
    for (int r = 0; r < 8; ++r)
      t1Buf[(mh + r) * PITCH + nloc] = f2bf(silu(c[r]));
  }
  __syncthreads();
  {
    v8f c;
    float bv = bn2[nloc];
#pragma unroll
    for (int r = 0; r < 8; ++r) c[r] = bv;
#pragma unroll
    for (int ks = 0; ks < 4; ++ks) {          // K = 128
      v16bf a  = load_afrag(t1Buf, PITCH, lane, ks);
      v16bf bw = load_bfrag(Wn2T, 128, lane, n0, ks);
      c = wmma_bf16(a, bw, c);
    }
#pragma unroll
    for (int r = 0; r < 8; ++r) {
      int gi = i0 + mh + r;
      outH[(b * NN + gi) * HH + nloc] = h[(b * NN + gi) * HH + nloc] + c[r];
    }
  }
  if (tid < 16) {
    int il = tid, gi = i0 + il;
    float cnt = fmaxf(neighS[il], 1.0f);
#pragma unroll
    for (int cc = 0; cc < 3; ++cc) {
      outX[(b * NN + gi) * 3 + cc] =
          x[(b * NN + gi) * 3 + cc] + coordAcc[il * 3 + cc] / cnt;
    }
  }
}

// ---------------------------------------------------------------------------
// Launch
// ---------------------------------------------------------------------------
extern "C" void kernel_launch(void* const* d_in, const int* in_sizes, int n_in,
                              void* d_out, int out_size, void* d_ws, size_t ws_size,
                              hipStream_t stream) {
  (void)in_sizes; (void)n_in; (void)out_size; (void)ws_size;
  const float* h   = (const float*)d_in[0];
  const float* x   = (const float*)d_in[1];
  const float* We1 = (const float*)d_in[2];
  const float* be1 = (const float*)d_in[3];
  const float* We2 = (const float*)d_in[4];
  const float* be2 = (const float*)d_in[5];
  const float* Wc1 = (const float*)d_in[6];
  const float* bc1 = (const float*)d_in[7];
  const float* Wc2 = (const float*)d_in[8];
  const float* bc2 = (const float*)d_in[9];
  const float* Wn1 = (const float*)d_in[10];
  const float* bn1 = (const float*)d_in[11];
  const float* Wn2 = (const float*)d_in[12];
  const float* bn2 = (const float*)d_in[13];
  const int* radius = (const int*)d_in[14];

  char* ws = (char*)d_ws;
  float*  Ai   = (float*)ws;                         // 2048*128 f32 = 1 MB
  float*  Bj   = (float*)(ws + (1 << 20));           // 1 MB
  __bf16* We2T = (__bf16*)(ws + (2 << 20));          // 128*128 bf16
  __bf16* Wc1T = We2T + 128 * 128;
  __bf16* Wn1T = Wc1T + 128 * 128;                   // 128*256 bf16
  __bf16* Wn2T = Wn1T + 128 * 256;

  float* outH = (float*)d_out;
  float* outX = outH + BB * NN * HH;

  egnn_pre_embed<<<BB * NN, 128, 0, stream>>>(h, We1, be1, Ai, Bj);
  egnn_pack_weights<<<320, 256, 0, stream>>>(We2, Wc1, Wn1, Wn2,
                                             We2T, Wc1T, Wn1T, Wn2T);
  egnn_edge<<<BB * (NN / 16), 256, 0, stream>>>(h, x, We1, be2, bc1, Wc2, bc2,
                                                bn1, bn2, Ai, Bj,
                                                We2T, Wc1T, Wn1T, Wn2T,
                                                radius, outH, outX);
}
